// BilateralSolverLocal_78821239816769
// MI455X (gfx1250) — matrix operand order repaired and checked by reading, not compiled, not run on hardware
//
#include <hip/hip_runtime.h>
#include <hip/hip_bf16.h>
#include <stdint.h>

#define H 448
#define W 448
#define HW (H*W)
#define KD 21
#define CHALF 10
#define OUTD 428
#define NBATCH 16
#define NCH 19
#define CLSID 17
#define KF 64        // padded feature depth: 48 real channels + 16 zeros
#define NTILE 27     // ceil(428/16)

typedef __attribute__((ext_vector_type(16))) __bf16 v16bf;
typedef __attribute__((ext_vector_type(8)))  float  v8f;

struct Q2 { uint4 a, b; };

static __device__ __forceinline__ v16bf load_frag(const uint4* p, int i0, int i1) {
  Q2 q; q.a = p[i0]; q.b = p[i1];
  return __builtin_bit_cast(v16bf, q);
}

// ---------------- init accumulators ----------------
__global__ void k_init(float* acc) {
  if (threadIdx.x < 2) acc[threadIdx.x] = 0.0f;
}

// ---------------- pass 1: softmax over 19 classes, p17 + batch-sum s --------
__global__ void k_softmax(const float* __restrict__ logits,
                          float* __restrict__ p17, float* __restrict__ s) {
  int idx = blockIdx.x * blockDim.x + threadIdx.x;
  if (idx >= HW) return;
  float ssum = 0.0f;
  for (int b = 0; b < NBATCH; ++b) {
    const float* base = logits + (size_t)b * NCH * HW + idx;
    float v[NCH];
    float mx = -1e30f;
#pragma unroll
    for (int c = 0; c < NCH; ++c) { v[c] = base[(size_t)c * HW]; mx = fmaxf(mx, v[c]); }
    float se = 0.0f;
#pragma unroll
    for (int c = 0; c < NCH; ++c) se += __expf(v[c] - mx);
    float p = __expf(v[CLSID] - mx) / se;
    p17[(size_t)b * HW + idx] = p;
    ssum += p;
  }
  s[idx] = ssum;
}

// ---------------- pass 2: bf16 feature pack (K padded to 64) + energy -------
__global__ void k_pack(const float* __restrict__ guide,
                       unsigned short* __restrict__ F, float* __restrict__ E) {
  int idx = blockIdx.x * blockDim.x + threadIdx.x;
  if (idx >= HW) return;
  unsigned short* fp = F + (size_t)idx * KF;
  float e = 0.0f;
#pragma unroll
  for (int m = 0; m < 48; ++m) {
    float f = guide[(size_t)m * HW + idx];            // m = b*3 + c layout
    unsigned u = __builtin_bit_cast(unsigned, f);
    unsigned r = u + 0x7FFFu + ((u >> 16) & 1u);      // RNE f32 -> bf16
    unsigned short hb = (unsigned short)(r >> 16);
    fp[m] = hb;
    float fb = __builtin_bit_cast(float, (unsigned)hb << 16);
    e += fb * fb;
  }
#pragma unroll
  for (int m = 48; m < KF; ++m) fp[m] = 0;
  E[idx] = e;
}

// ---------------- loss2: per-(b,w) column reduction over h ------------------
__global__ void k_loss2(const float* __restrict__ p17, const int* __restrict__ tgt,
                        float* __restrict__ acc) {
  int idx = blockIdx.x * blockDim.x + threadIdx.x;   // b*W + w
  if (idx >= NBATCH * W) return;
  int b = idx / W, w = idx - b * W;
  const float* pp = p17 + (size_t)b * HW + w;
  const int*   tp = tgt + (size_t)b * HW + w;
  float sp = 0.0f, spt = 0.0f; int n1 = 0;
  for (int h = 0; h < H; ++h) {
    float p = pp[(size_t)h * W];
    int t = (tp[(size_t)h * W] == CLSID);
    sp += p; n1 += t; spt += t ? p : 0.0f;
  }
  // logsumexp over a {0,1} column of length 448: log(448 + n1*(e-1))
  float lse = __logf(448.0f + (float)n1 * 1.7182818284590452f);
  atomicAdd(acc + 1, spt - lse * sp);
}

// ---------------- main bilateral accumulation via WMMA ----------------------
// ssd(o,p) = E_c + E_n - 2 * <feat_c, feat_n>, cross term on v_wmma bf16.
// LDS stages rows [y, y+20] of E and s (75 KB dynamic LDS; gfx1250 WGP = 320 KB).
__global__ void __launch_bounds__(256) k_bilateral(
    const unsigned short* __restrict__ F, const float* __restrict__ E,
    const float* __restrict__ s, float* __restrict__ acc) {
  extern __shared__ float smem[];
  float* Es = smem;              // [21][448]
  float* Ss = smem + KD * W;     // [21][448]

  const int y    = blockIdx.x;          // output row 0..427
  const int tid  = threadIdx.x;
  const int wave = tid >> 5;
  const int lane = tid & 31;
  const int l15  = lane & 15;
  const int hi   = (lane >> 4) & 1;
  const int ycen = y + CHALF;

  // ---- stage E and s rows into LDS (float4, fully coalesced) ----
  {
    const int nq = KD * (W / 4);             // 21*112 float4 per array
    for (int i = tid; i < nq; i += 256) {
      int r  = i / (W / 4);
      int c4 = i - r * (W / 4);
      ((float4*)Es)[i] = ((const float4*)(E + (size_t)(y + r) * W))[c4];
      ((float4*)Ss)[i] = ((const float4*)(s + (size_t)(y + r) * W))[c4];
    }
  }
  __syncthreads();

  // A-fragment lane chunks: lanes<16 -> K{0..7,16..23}, hi -> +8 (ushort units)
  const int ia = hi;          // uint4 index of first A chunk
  const int ib = hi;          // uint4 index of B chunk

  float accw = 0.0f;
  const int NU = KD * NTILE;  // 567 (oi, xtile) units per row
  for (int u = wave; u < NU; u += 8) {
    int oi  = u / NTILE;
    int tx  = u - oi * NTILE;
    int ynb = y + oi;
    int xc0 = CHALF + 16 * tx;                 // image col of center pixel m=0
    int mMax = (tx == NTILE - 1) ? 12 : 16;    // 428 = 26*16 + 12
    float di  = (float)(CHALF - oi);
    float spc = di * di;

    // A: 16 center pixels (row ycen), K = 0..63 in two fragments
    const uint4* ap = (const uint4*)(F + ((size_t)ycen * W + (xc0 + l15)) * KF);
    v16bf a0 = load_frag(ap, ia,     ia + 2);
    v16bf a1 = load_frag(ap, ia + 4, ia + 6);

    // prefetch next unit's A row (global_prefetch_b8)
    int un = u + 8;
    if (un < NU) {
      int oin = un / NTILE;
      int txn = un - oin * NTILE;
      __builtin_prefetch(F + ((size_t)ycen * W + (CHALF + 16 * txn + l15)) * KF, 0, 1);
      __builtin_prefetch(F + ((size_t)(y + oin) * W + (16 * txn + l15)) * KF, 0, 1);
    }

    // center-column E/s values: depend only on m = r + 8*hi (LDS broadcasts)
    float ec[8], sc[8];
#pragma unroll
    for (int r = 0; r < 8; ++r) {
      int xcc = xc0 + r + hi * 8;              // <= 441, always in-bounds
      ec[r] = Es[CHALF * W + xcc];
      sc[r] = Ss[CHALF * W + xcc];
    }

#pragma unroll
    for (int t = 0; t < 3; ++t) {
      int col  = xc0 - 16 + 16 * t + l15;      // neighbor pixel column (indep of m)
      int colc = min(max(col, 0), W - 1);      // clamp; invalid outputs masked below
      float en = Es[oi * W + colc];
      float sn = Ss[oi * W + colc];

      const uint4* bp = (const uint4*)(F + ((size_t)ynb * W + colc) * KF);
      v16bf b0 = load_frag(bp, ib,     ib + 2);
      v16bf b1 = load_frag(bp, ib + 4, ib + 6);

      v8f c = {};
      c = __builtin_amdgcn_wmma_f32_16x16x32_bf16(false, a0, false, b0,
                                                  (short)0, c, false, false);
      c = __builtin_amdgcn_wmma_f32_16x16x32_bf16(false, a1, false, b1,
                                                  (short)0, c, false, false);

      // branchless consume: D element (m = r+8*hi, n = l15), delta = 16t-16+n-m
#pragma unroll
      for (int r = 0; r < 8; ++r) {
        int m = r + hi * 8;
        int d = 16 * t - 16 + l15 - m;
        bool use = (d >= -10) & (d <= 10) & (m < mMax) & !((oi == CHALF) & (d == 0));
        float valid = use ? 1.0f : 0.0f;
        float ssd = fmaxf(ec[r] + en - 2.0f * c[r], 0.0f);
        float fd  = (float)d;
        // 16*position + reference = -(d^2 + (10-oi)^2 + ssd)/128
        float wgt = __expf(-(fd * fd + spc + ssd) * (1.0f / 128.0f));
        float dv  = sc[r] - sn;
        accw += valid * (wgt * dv * dv);
      }
    }
  }

  // wave32 butterfly reduction, one atomic per wave
#pragma unroll
  for (int off = 16; off > 0; off >>= 1) accw += __shfl_xor(accw, off, 32);
  if (lane == 0) atomicAdd(acc + 0, accw);
}

// ---------------- final scalar combine --------------------------------------
__global__ void k_final(const float* __restrict__ acc, float* __restrict__ out) {
  float loss1 = 128.0f * acc[0] / (440.0f * 428.0f * 428.0f);
  float loss2 = -acc[1] / (float)(NBATCH * W);
  out[0] = (loss1 * 300.0f + loss2) / 24.0f;
}

extern "C" void kernel_launch(void* const* d_in, const int* in_sizes, int n_in,
                              void* d_out, int out_size, void* d_ws, size_t ws_size,
                              hipStream_t stream) {
  (void)in_sizes; (void)n_in; (void)out_size; (void)ws_size;
  const float* logits = (const float*)d_in[0];   // (16,19,448,448) f32
  const float* guide  = (const float*)d_in[1];   // (16,3,448,448)  f32
  const int*   target = (const int*)d_in[2];     // (16,448,448)    i32
  float* out = (float*)d_out;

  char* ws = (char*)d_ws;
  size_t off = 0;
  auto carve = [&](size_t bytes) -> char* {
    char* p = ws + off;
    off = (off + bytes + 255) & ~(size_t)255;
    return p;
  };
  float*          acc = (float*)carve(2 * sizeof(float));
  float*          p17 = (float*)carve((size_t)NBATCH * HW * sizeof(float));
  float*          s   = (float*)carve((size_t)HW * sizeof(float));
  float*          E   = (float*)carve((size_t)HW * sizeof(float));
  unsigned short* F   = (unsigned short*)carve((size_t)HW * KF * sizeof(unsigned short));

  const size_t smem_bytes = (size_t)2 * KD * W * sizeof(float);  // 75264 B

  k_init<<<1, 32, 0, stream>>>(acc);
  k_softmax<<<(HW + 255) / 256, 256, 0, stream>>>(logits, p17, s);
  k_pack<<<(HW + 255) / 256, 256, 0, stream>>>(guide, F, E);
  k_loss2<<<(NBATCH * W + 255) / 256, 256, 0, stream>>>(p17, target, acc);
  k_bilateral<<<OUTD, 256, smem_bytes, stream>>>(F, E, s, acc);
  k_final<<<1, 1, 0, stream>>>(acc, out);
}
